// CausalSelfAttention_70102456205661
// MI455X (gfx1250) — matrix operand order repaired and checked by reading, not compiled, not run on hardware
//
#include <hip/hip_runtime.h>

#define BB 2
#define TT 2048
#define CC 1024
#define HH 16
#define DD 64
#define NQKV (3 * CC)

typedef __attribute__((ext_vector_type(16))) __bf16 v16bf;
typedef __attribute__((ext_vector_type(8)))  __bf16 v8bf;
typedef __attribute__((ext_vector_type(4)))  __bf16 v4bf;
typedef __attribute__((ext_vector_type(8)))  float  v8f;
typedef __attribute__((ext_vector_type(4)))  int    v4i;

// ---------------- CDNA5 async Global->LDS copy (ASYNCcnt-tracked) ----------------
#if defined(__has_builtin)
#  if __has_builtin(__builtin_amdgcn_global_load_async_to_lds_b128)
#    define HAVE_ASYNC_BUILTIN 1
#  endif
#  if __has_builtin(__builtin_amdgcn_s_wait_asynccnt)
#    define HAVE_WAITASYNC_BUILTIN 1
#  endif
#endif

typedef __attribute__((address_space(1))) v4i* gv4ip;   // global (AS1) int4*
typedef __attribute__((address_space(3))) v4i* lv4ip;   // LDS (AS3) int4*

static __device__ inline void async_cp16(const __bf16* g, __bf16* l) {
  // generic LDS address carries the LDS byte offset in bits [31:0]
  unsigned lds_off = (unsigned)(size_t)l;
#ifdef HAVE_ASYNC_BUILTIN
  __builtin_amdgcn_global_load_async_to_lds_b128(
      (gv4ip)(size_t)g, (lv4ip)lds_off, 0, 0);
#else
  asm volatile("global_load_async_to_lds_b128 %0, %1, off"
               :: "v"(lds_off), "v"(g) : "memory");
#endif
}

static __device__ inline void wait_async_le8() {
#ifdef HAVE_WAITASYNC_BUILTIN
  __builtin_amdgcn_s_wait_asynccnt(8);
#else
  asm volatile("s_wait_asynccnt 8" ::: "memory");
#endif
}
static __device__ inline void wait_async_0() {
#ifdef HAVE_WAITASYNC_BUILTIN
  __builtin_amdgcn_s_wait_asynccnt(0);
#else
  asm volatile("s_wait_asynccnt 0" ::: "memory");
#endif
}

static __device__ inline v16bf combine16(v8bf lo, v8bf hi) {
  return __builtin_shufflevector(lo, hi, 0, 1, 2, 3, 4, 5, 6, 7, 8, 9, 10, 11, 12, 13, 14, 15);
}

// ---- WMMA bf16 fragment loaders from a row-chunk base pointer ----
// A: 16x32, lane&15 = M row, lane>>4 -> K halves {0..7,16..23} / {8..15,24..31}
static __device__ inline v16bf frag_a(const __bf16* rowbase) {
  return combine16(*(const v8bf*)rowbase, *(const v8bf*)(rowbase + 16));
}
// B (N-major rows): lane&15 = N col, lane>>4 -> K halves {0..7,16..23} / {8..15,24..31}
static __device__ inline v16bf frag_b(const __bf16* rowbase) {
  return combine16(*(const v8bf*)rowbase, *(const v8bf*)(rowbase + 16));
}

// shared-tile layout: [buf][A=0/B=1][row 0..127][col 0..31]
static __device__ inline int sidx(int buf, int ab, int row, int col) {
  return ((buf * 2 + ab) * 128 + row) * 32 + col;
}

// ---------------- Pack kernels (one-time f32 -> bf16) ----------------
__global__ __launch_bounds__(256) void cvt_bf16_kernel(const float* __restrict__ in,
                                                       __bf16* __restrict__ out) {
  const int i = (blockIdx.x * 256 + threadIdx.x) * 4;
  float4 v = *(const float4*)(in + i);
  v4bf o = {(__bf16)v.x, (__bf16)v.y, (__bf16)v.z, (__bf16)v.w};
  *(v4bf*)(out + i) = o;
}

// in: K x N (row-major), out: N x K bf16
__global__ __launch_bounds__(256) void transpose_cvt_kernel(const float* __restrict__ in,
                                                            __bf16* __restrict__ out,
                                                            int K, int N) {
  __shared__ float tile[32][33];
  const int tx = threadIdx.x & 31;
  const int ty = threadIdx.x >> 5;
  const int n0 = blockIdx.x * 32;
  const int k0 = blockIdx.y * 32;
#pragma unroll
  for (int r = 0; r < 4; ++r)
    tile[ty + r * 8][tx] = in[(size_t)(k0 + ty + r * 8) * N + n0 + tx];
  __syncthreads();
#pragma unroll
  for (int r = 0; r < 4; ++r)
    out[(size_t)(n0 + ty + r * 8) * K + k0 + tx] = (__bf16)tile[tx][ty + r * 8];
}

// ---------------- Kernel 1: QKV GEMM (128x128 block, async-LDS staged) + RoPE ----------------
__global__ __launch_bounds__(128) void qkv_gemm_rope_kernel(
    const __bf16* __restrict__ xb, const __bf16* __restrict__ wqT,
    const float* __restrict__ fcos, const float* __restrict__ fsin,
    __bf16* __restrict__ qb, __bf16* __restrict__ kbf, __bf16* __restrict__ vT,
    float* __restrict__ outK, float* __restrict__ outV) {
  __shared__ __align__(16) __bf16 smem[2 * 2 * 128 * 32];   // 32 KB, double-buffered A/B tiles
  const int tid  = threadIdx.x;
  const int lane = tid & 31;
  const int wave = tid >> 5;
  const int wm = wave >> 1, wn = wave & 1;                  // 2x2 waves of 64x64
  const int NB = NQKV / 128;                                // 24
  const int m0blk = (blockIdx.x / NB) * 128;
  const int n0blk = (blockIdx.x % NB) * 128;

  const int am  = lane & 15;
  const int akb = (lane >> 4) << 3;                         // 0/8
  const int bn  = lane & 15;
  const int bkg = (lane >> 4) << 3;                         // 0/8

  // async stage one 128x32 A slice + 128x32 B slice
  auto load_tile = [&](int kt, int buf) {
    const __bf16* ga = xb  + (size_t)(m0blk + tid) * CC + kt * 32;
    const __bf16* gb = wqT + (size_t)(n0blk + tid) * CC + kt * 32;
#pragma unroll
    for (int c = 0; c < 4; ++c) {
      async_cp16(ga + c * 8, smem + sidx(buf, 0, tid, c * 8));
      async_cp16(gb + c * 8, smem + sidx(buf, 1, tid, c * 8));
    }
  };

  v8f acc[4][4] = {};
  load_tile(0, 0);
  const int KT = CC / 32;                                   // 32
  for (int kt = 0; kt < KT; ++kt) {
    const int cur = kt & 1;
    if (kt + 1 < KT) { load_tile(kt + 1, cur ^ 1); wait_async_le8(); }
    else             { wait_async_0(); }
    __syncthreads();
    v16bf a[4], b[4];
#pragma unroll
    for (int mi = 0; mi < 4; ++mi)
      a[mi] = frag_a(smem + sidx(cur, 0, wm * 64 + mi * 16 + am, akb));
#pragma unroll
    for (int ni = 0; ni < 4; ++ni)
      b[ni] = frag_b(smem + sidx(cur, 1, wn * 64 + ni * 16 + bn, bkg));
#pragma unroll
    for (int mi = 0; mi < 4; ++mi)
#pragma unroll
      for (int ni = 0; ni < 4; ++ni)
        acc[mi][ni] = __builtin_amdgcn_wmma_f32_16x16x32_bf16(false, a[mi], false, b[ni],
                                                              (short)0, acc[mi][ni], false, false);
    __syncthreads();
  }

  // Epilogue: wave's 64x64 tile covers exactly one head of one region
  const int m0 = m0blk + wm * 64;
  const int n0 = n0blk + wn * 64;
  const int hi = lane >> 4;
  const int nl = lane & 15;
  const int region = n0 / CC;                               // 0=Q 1=K 2=V
  const int h = (n0 % CC) / DD;
#pragma unroll
  for (int ni = 0; ni < 4; ++ni) {
    const int d = ni * 16 + nl;
    const int pair = d >> 1;
#pragma unroll
    for (int mi = 0; mi < 4; ++mi) {
#pragma unroll
      for (int r = 0; r < 8; ++r) {
        const int row = m0 + mi * 16 + r + 8 * hi;
        const int b_ = row / TT;
        const int t_ = row % TT;
        float v = acc[mi][ni][r];
        float out;
        if (region < 2) {
          float p = __shfl_xor(v, 1, 32);                   // RoPE partner column
          float c = fcos[t_ * (DD / 2) + pair];
          float s = fsin[t_ * (DD / 2) + pair];
          out = ((d & 1) == 0) ? (v * c - p * s) : (p * s + v * c);
        } else {
          out = v;
        }
        const size_t idx = (((size_t)(b_ * TT + t_) * HH) + h) * DD + d;  // [B,T,H,D]
        if (region == 0) {
          qb[idx] = (__bf16)out;
        } else if (region == 1) {
          outK[idx] = out;
          kbf[idx] = (__bf16)out;
        } else {
          outV[idx] = out;
          vT[(((size_t)(b_ * HH + h) * DD) + d) * TT + t_] = (__bf16)out;  // [B,H,D,T]
        }
      }
    }
  }
}

// ---------------- Kernel 2: flash-style causal attention ----------------
__global__ __launch_bounds__(128) void attn_kernel(
    const __bf16* __restrict__ qb, const __bf16* __restrict__ kbf,
    const __bf16* __restrict__ vT, __bf16* __restrict__ attb) {
  __shared__ __align__(16) __bf16 p_lds[4][16][32];
  const int lane = threadIdx.x & 31;
  const int wave = threadIdx.x >> 5;
  const int QT = TT / 16;
  const int job = blockIdx.x * 4 + wave;
  const int b_ = job / (HH * QT);
  const int rem = job % (HH * QT);
  const int h = rem / QT;
  const int q0 = (rem % QT) * 16;

  const int am  = lane & 15;
  const int akb = (lane >> 4) << 3;
  const int hi  = lane >> 4;
  const int ncol = lane & 15;
  const int bkb = (lane >> 4) << 3;
  const size_t rstr = (size_t)HH * DD;

  const __bf16* qrow = qb + (((size_t)(b_ * TT + q0 + am) * HH) + h) * DD;
  v16bf aq0 = combine16(*(const v8bf*)(qrow + akb), *(const v8bf*)(qrow + akb + 16));
  v16bf aq1 = combine16(*(const v8bf*)(qrow + 32 + akb), *(const v8bf*)(qrow + 32 + akb + 16));

  float mrow[8], lrow[8];
  v8f o0 = {}, o1 = {}, o2 = {}, o3 = {};
#pragma unroll
  for (int r = 0; r < 8; ++r) { mrow[r] = -1e30f; lrow[r] = 0.0f; }

  const int nblocks = (q0 + 16 + 31) / 32;
  for (int kb = 0; kb < nblocks; ++kb) {
    const int key0 = kb * 32;
    v8f s0 = {}, s1 = {};
#pragma unroll
    for (int c = 0; c < 2; ++c) {
      const __bf16* kp0 = kbf + (((size_t)(b_ * TT + key0 + ncol) * HH) + h) * DD + c * 32 + bkb;
      const __bf16* kp1 = kp0 + 16 * rstr;
      v16bf bk0 = combine16(*(const v8bf*)kp0, *(const v8bf*)(kp0 + 16));
      v16bf bk1 = combine16(*(const v8bf*)kp1, *(const v8bf*)(kp1 + 16));
      v16bf aq = (c == 0) ? aq0 : aq1;
      s0 = __builtin_amdgcn_wmma_f32_16x16x32_bf16(false, aq, false, bk0, (short)0, s0, false, false);
      s1 = __builtin_amdgcn_wmma_f32_16x16x32_bf16(false, aq, false, bk1, (short)0, s1, false, false);
    }

    float p0[8], p1[8];
#pragma unroll
    for (int r = 0; r < 8; ++r) {
      const int qi  = q0 + r + 8 * hi;
      const int k0i = key0 + ncol;
      s0[r] = (k0i      <= qi) ? s0[r] * 0.125f : -1e30f;
      s1[r] = (k0i + 16 <= qi) ? s1[r] * 0.125f : -1e30f;

      float mx = fmaxf(s0[r], s1[r]);
#pragma unroll
      for (int off = 1; off < 16; off <<= 1) mx = fmaxf(mx, __shfl_xor(mx, off, 32));
      const float mnew = fmaxf(mrow[r], mx);
      const float alpha = __expf(mrow[r] - mnew);
      p0[r] = __expf(s0[r] - mnew);
      p1[r] = __expf(s1[r] - mnew);
      float sum = p0[r] + p1[r];
#pragma unroll
      for (int off = 1; off < 16; off <<= 1) sum += __shfl_xor(sum, off, 32);
      lrow[r] = lrow[r] * alpha + sum;
      mrow[r] = mnew;
      o0[r] *= alpha; o1[r] *= alpha; o2[r] *= alpha; o3[r] *= alpha;
    }

#pragma unroll
    for (int r = 0; r < 8; ++r) {
      const int m = r + 8 * hi;
      p_lds[wave][m][ncol]      = (__bf16)p0[r];
      p_lds[wave][m][ncol + 16] = (__bf16)p1[r];
    }
    const __bf16* pr = &p_lds[wave][am][0];
    v16bf ap = combine16(*(const v8bf*)(pr + akb), *(const v8bf*)(pr + akb + 16));

#pragma unroll
    for (int dt = 0; dt < 4; ++dt) {
      const __bf16* vp = vT + (((size_t)(b_ * HH + h) * DD) + dt * 16 + ncol) * TT + key0 + bkb;
      v16bf bv = combine16(*(const v8bf*)vp, *(const v8bf*)(vp + 16));
      v8f* op = (dt == 0) ? &o0 : (dt == 1) ? &o1 : (dt == 2) ? &o2 : &o3;
      *op = __builtin_amdgcn_wmma_f32_16x16x32_bf16(false, ap, false, bv, (short)0, *op, false, false);
    }
  }

#pragma unroll
  for (int r = 0; r < 8; ++r) {
    const float inv = 1.0f / lrow[r];
    const int t_ = q0 + r + 8 * hi;
    const size_t base = (((size_t)(b_ * TT + t_) * HH) + h) * DD;
    attb[base + 0 * 16 + ncol] = (__bf16)(o0[r] * inv);
    attb[base + 1 * 16 + ncol] = (__bf16)(o1[r] * inv);
    attb[base + 2 * 16 + ncol] = (__bf16)(o2[r] * inv);
    attb[base + 3 * 16 + ncol] = (__bf16)(o3[r] * inv);
  }
}

// ---------------- Kernel 3: projection GEMM (128x128 block, async-LDS staged) ----------------
__global__ __launch_bounds__(128) void proj_gemm_kernel(
    const __bf16* __restrict__ attb, const __bf16* __restrict__ wpT,
    float* __restrict__ y) {
  __shared__ __align__(16) __bf16 smem[2 * 2 * 128 * 32];
  const int tid  = threadIdx.x;
  const int lane = tid & 31;
  const int wave = tid >> 5;
  const int wm = wave >> 1, wn = wave & 1;
  const int NB = CC / 128;                                  // 8
  const int m0blk = (blockIdx.x / NB) * 128;
  const int n0blk = (blockIdx.x % NB) * 128;

  const int am  = lane & 15;
  const int akb = (lane >> 4) << 3;
  const int bn  = lane & 15;
  const int bkg = (lane >> 4) << 3;

  auto load_tile = [&](int kt, int buf) {
    const __bf16* ga = attb + (size_t)(m0blk + tid) * CC + kt * 32;
    const __bf16* gb = wpT  + (size_t)(n0blk + tid) * CC + kt * 32;
#pragma unroll
    for (int c = 0; c < 4; ++c) {
      async_cp16(ga + c * 8, smem + sidx(buf, 0, tid, c * 8));
      async_cp16(gb + c * 8, smem + sidx(buf, 1, tid, c * 8));
    }
  };

  v8f acc[4][4] = {};
  load_tile(0, 0);
  const int KT = CC / 32;
  for (int kt = 0; kt < KT; ++kt) {
    const int cur = kt & 1;
    if (kt + 1 < KT) { load_tile(kt + 1, cur ^ 1); wait_async_le8(); }
    else             { wait_async_0(); }
    __syncthreads();
    v16bf a[4], b[4];
#pragma unroll
    for (int mi = 0; mi < 4; ++mi)
      a[mi] = frag_a(smem + sidx(cur, 0, wm * 64 + mi * 16 + am, akb));
#pragma unroll
    for (int ni = 0; ni < 4; ++ni)
      b[ni] = frag_b(smem + sidx(cur, 1, wn * 64 + ni * 16 + bn, bkg));
#pragma unroll
    for (int mi = 0; mi < 4; ++mi)
#pragma unroll
      for (int ni = 0; ni < 4; ++ni)
        acc[mi][ni] = __builtin_amdgcn_wmma_f32_16x16x32_bf16(false, a[mi], false, b[ni],
                                                              (short)0, acc[mi][ni], false, false);
    __syncthreads();
  }

  const int m0 = m0blk + wm * 64;
  const int n0 = n0blk + wn * 64;
  const int hi = lane >> 4;
  const int nl = lane & 15;
#pragma unroll
  for (int ni = 0; ni < 4; ++ni) {
    const int n = n0 + ni * 16 + nl;
#pragma unroll
    for (int mi = 0; mi < 4; ++mi)
#pragma unroll
      for (int r = 0; r < 8; ++r) {
        const int row = m0 + mi * 16 + r + 8 * hi;
        y[(size_t)row * CC + n] = acc[mi][ni][r];
      }
  }
}

extern "C" void kernel_launch(void* const* d_in, const int* in_sizes, int n_in,
                              void* d_out, int out_size, void* d_ws, size_t ws_size,
                              hipStream_t stream) {
  const float* x     = (const float*)d_in[0];
  const float* fcos  = (const float*)d_in[1];
  const float* fsin  = (const float*)d_in[2];
  const float* wqkv  = (const float*)d_in[3];
  const float* wproj = (const float*)d_in[4];

  const size_t NELT = (size_t)BB * TT * CC;
  float* y    = (float*)d_out;
  float* outK = y + NELT;
  float* outV = outK + NELT;

  __bf16* p = (__bf16*)d_ws;
  __bf16* xb   = p; p += NELT;
  __bf16* wqT  = p; p += (size_t)NQKV * CC;
  __bf16* wpT  = p; p += (size_t)CC * CC;
  __bf16* qb   = p; p += NELT;
  __bf16* kbf  = p; p += NELT;
  __bf16* vT   = p; p += NELT;
  __bf16* attb = p; p += NELT;

  cvt_bf16_kernel<<<(int)(NELT / 1024), 256, 0, stream>>>(x, xb);
  transpose_cvt_kernel<<<dim3(NQKV / 32, CC / 32), 256, 0, stream>>>(wqkv, wqT, CC, NQKV);
  transpose_cvt_kernel<<<dim3(CC / 32, CC / 32), 256, 0, stream>>>(wproj, wpT, CC, CC);

  // QKV: (4096/128)*(3072/128) = 768 blocks of 128 threads
  qkv_gemm_rope_kernel<<<768, 128, 0, stream>>>(xb, wqT, fcos, fsin, qb, kbf, vT, outK, outV);
  // Attention: B*H*(T/16) = 4096 wave jobs, 4/block
  attn_kernel<<<4096 / 4, 128, 0, stream>>>(qb, kbf, vT, attb);
  // Projection: (4096/128)*(1024/128) = 256 blocks
  proj_gemm_kernel<<<256, 128, 0, stream>>>(attb, wpT, y);
}
